// TPRencoder_43903155700270
// MI455X (gfx1250) — matrix operand order repaired and checked by reading, not compiled, not run on hardware
//
#include <hip/hip_runtime.h>

// ---------------------------------------------------------------------------
// Types for CDNA5 WMMA (wave32): v_wmma_f32_16x16x32_bf16
// ---------------------------------------------------------------------------
typedef __bf16 bf16;
typedef __attribute__((ext_vector_type(16))) __bf16 v16bf;
typedef __attribute__((ext_vector_type(8)))  __bf16 v8bf;
typedef __attribute__((ext_vector_type(8)))  float  v8f;
typedef __attribute__((ext_vector_type(4)))  unsigned int u32x4;
typedef __attribute__((ext_vector_type(4)))  int i32x4;
typedef __attribute__((ext_vector_type(8)))  int i32x8;

union V16U { v16bf v; v8bf h[2]; };

#define Bsz   64
#define Ssz   256
#define INDIM 1024
#define HIDN  1024
#define KDIM  2048   // concat(x_t, T_prev)
#define NG    8192   // 4H for F + 4H for R
#define NSYM  512
#define NROLE 256
#define DSYM  32
#define DROLE 32
#define TILEW 256    // K-columns staged per TDM phase (64x256 bf16 = 32KB)

__device__ inline bf16 to_bf16(float f) { return (bf16)f; }

__device__ inline v8f wmma_bf16(v16bf a, v16bf b, v8f c) {
  return __builtin_amdgcn_wmma_f32_16x16x32_bf16(false, a, false, b, (short)0, c,
                                                 false, false);
}

// A fragment (16x32 bf16) from an LDS tile laid out row-major with stride TILEW.
__device__ inline v16bf load_a_frag_lds(const bf16* tile, int r, int k0, int half) {
  const bf16* p = tile + r * TILEW + k0;
  V16U u;
  u.h[0] = *(const v8bf*)(p + half * 8);
  u.h[1] = *(const v8bf*)(p + 16 + half * 8);
  return u.v;
}

// A fragment directly from global, row stride ldk.
__device__ inline v16bf load_a_frag_g(const bf16* A, int m0, int kk, int ldk, int lane) {
  int r = m0 + (lane & 15);
  int half = lane >> 4;
  const bf16* p = A + (size_t)r * ldk + kk;
  V16U u;
  u.h[0] = *(const v8bf*)(p + half * 8);
  u.h[1] = *(const v8bf*)(p + 16 + half * 8);
  return u.v;
}

// B fragment (32x16): weight stored as W[N][K] row-major; B[k][n] = W[n][k].
__device__ inline v16bf load_b_frag(const bf16* W, int n0, int kk, int ldk, int lane) {
  int col = n0 + (lane & 15);
  int half = lane >> 4;
  const bf16* p = W + (size_t)col * ldk + kk;
  V16U u;
  u.h[0] = *(const v8bf*)(p + half * 8);
  u.h[1] = *(const v8bf*)(p + 16 + half * 8);
  return u.v;
}

__device__ inline float sigf(float x) { return 1.0f / (1.0f + __expf(-x)); }

// ---------------------------------------------------------------------------
// Tensor Data Mover: async 2D tile (64 rows x TILEW cols, bf16) global -> LDS
// ---------------------------------------------------------------------------
#if __has_builtin(__builtin_amdgcn_tensor_load_to_lds)
#define HAVE_TDM 1
#else
#define HAVE_TDM 0
#endif

#if HAVE_TDM
__device__ inline void tdm_load_2d(unsigned lds_addr, const void* gsrc,
                                   unsigned tileW, unsigned rows,
                                   unsigned tensorW, unsigned strideElems) {
  unsigned long long ga = (unsigned long long)(uintptr_t)gsrc;
  u32x4 g0;
  g0.x = 1u;                                    // count = 1 valid descriptor
  g0.y = lds_addr;                              // LDS byte address
  g0.z = (unsigned)ga;                          // global addr [31:0]
  g0.w = (unsigned)((ga >> 32) & 0x01FFFFFFull) // global addr [56:32]
         | (2u << 30);                          // type = 2 ("image")
  i32x8 g1;
  g1[0] = (int)(1u << 16);                                  // data_size=1 -> 2 bytes
  g1[1] = (int)((tensorW & 0xFFFFu) << 16);                 // tensor_dim0 lo16
  g1[2] = (int)((tensorW >> 16) & 0xFFFFu)                  // tensor_dim0 hi16
        | (int)((rows & 0xFFFFu) << 16);                    // tensor_dim1 lo16
  g1[3] = (int)((rows >> 16) & 0xFFFFu)                     // tensor_dim1 hi16
        | (int)((tileW & 0xFFFFu) << 16);                   // tile_dim0
  g1[4] = (int)(rows & 0xFFFFu);                            // tile_dim1 (tile_dim2=0)
  g1[5] = (int)strideElems;                                 // tensor_dim0_stride lo32
  g1[6] = 0;
  g1[7] = 0;
  i32x4 z4 = {};
#if defined(__clang_major__) && __clang_major__ >= 23
  i32x8 z8 = {};
  __builtin_amdgcn_tensor_load_to_lds(g0, g1, z4, z4, z8, 0);
#else
  __builtin_amdgcn_tensor_load_to_lds(g0, g1, z4, z4, 0);
#endif
}
#endif

// Stage a 64 x TILEW tile (row stride `strideElems` in global) into LDS.
__device__ inline void tile_issue(bf16* dst, const bf16* src, int strideElems,
                                  int tid, int wave) {
#if HAVE_TDM
  if (wave == 0)
    tdm_load_2d((unsigned)(uintptr_t)dst, src, (unsigned)TILEW, 64u,
                (unsigned)strideElems, (unsigned)strideElems);
  (void)tid;
#else
  for (int i = tid; i < 64 * (TILEW / 8); i += 256) {
    int row = i / (TILEW / 8), c8 = (i % (TILEW / 8)) * 8;
    *(v8bf*)(dst + row * TILEW + c8) = *(const v8bf*)(src + (size_t)row * strideElems + c8);
  }
  (void)wave;
#endif
}

__device__ inline void tile_wait() {
#if HAVE_TDM
  __builtin_amdgcn_s_wait_tensorcnt(0);
#endif
}

// ---------------------------------------------------------------------------
// Prep kernels (run once per call)
// ---------------------------------------------------------------------------
__global__ void k_prep_x(const float* __restrict__ x, bf16* __restrict__ xb) {
  const size_t total = (size_t)Bsz * Ssz * INDIM;
  for (size_t i = (size_t)blockIdx.x * blockDim.x + threadIdx.x; i < total;
       i += (size_t)gridDim.x * blockDim.x) {
    size_t k = i & 1023;
    size_t b = (i >> 10) & 63;
    size_t t = i >> 16;
    xb[i] = to_bf16(x[(((b << 8) + t) << 10) + k]);
  }
}

__global__ void k_prep_w(const float* __restrict__ WihF, const float* __restrict__ WhhF,
                         const float* __restrict__ WihR, const float* __restrict__ WhhR,
                         bf16* __restrict__ Wcomb) {
  const size_t total = (size_t)NG * KDIM;
  for (size_t i = (size_t)blockIdx.x * blockDim.x + threadIdx.x; i < total;
       i += (size_t)gridDim.x * blockDim.x) {
    size_t n = i >> 11, k = i & 2047;
    float v;
    if (n < 4096) v = (k < 1024) ? WihF[n * 1024 + k] : WhhF[n * 1024 + (k - 1024)];
    else {
      size_t n2 = n - 4096;
      v = (k < 1024) ? WihR[n2 * 1024 + k] : WhhR[n2 * 1024 + (k - 1024)];
    }
    Wcomb[i] = to_bf16(v);
  }
}

__global__ void k_prep_small(const float* bihF, const float* bhhF,
                             const float* bihR, const float* bhhR,
                             const float* WaF_w, const float* WaR_w,
                             const float* F_w, const float* R_w,
                             float* biasComb, bf16* WaFb, bf16* WaRb,
                             bf16* Fwb, bf16* Rwb) {
  size_t i0 = (size_t)blockIdx.x * blockDim.x + threadIdx.x;
  size_t st = (size_t)gridDim.x * blockDim.x;
  for (size_t i = i0; i < NG; i += st)
    biasComb[i] = (i < 4096) ? (bihF[i] + bhhF[i]) : (bihR[i - 4096] + bhhR[i - 4096]);
  for (size_t i = i0; i < (size_t)NSYM * HIDN; i += st) WaFb[i] = to_bf16(WaF_w[i]);
  for (size_t i = i0; i < (size_t)NROLE * HIDN; i += st) WaRb[i] = to_bf16(WaR_w[i]);
  for (size_t i = i0; i < (size_t)DSYM * NSYM; i += st) Fwb[i] = to_bf16(F_w[i]);
  for (size_t i = i0; i < (size_t)DROLE * NROLE; i += st) Rwb[i] = to_bf16(R_w[i]);
}

// ---------------------------------------------------------------------------
// Step kernel 1: gates[64,8192] = [x_t | T_prev] @ Wcomb^T + biasComb
// 64 blocks x 256 threads. 8 TDM phases of K=256, double-buffered.
// Per k-chunk: load all 4 A-fragments into distinct regs, then 4 WMMAs
// back-to-back (lets the scheduler use counted dscnt waits, not full drains).
// ---------------------------------------------------------------------------
__global__ __launch_bounds__(256) void k_gates(const bf16* __restrict__ xb_t,
                                               const bf16* __restrict__ Tprev,
                                               const bf16* __restrict__ Wcomb,
                                               const float* __restrict__ biasComb,
                                               float* __restrict__ gates) {
  __shared__ bf16 Abuf[2][64 * TILEW];     // 2 x 32KB
  const int tid = threadIdx.x, lane = tid & 31, wave = tid >> 5;
  const int nBase = blockIdx.x * 128 + wave * 16;
  const int r0 = lane & 15, half = lane >> 4;
  v8f acc[4] = {};

  tile_issue(Abuf[0], xb_t, 1024, tid, wave);          // phase 0 (kk = 0..255)
  for (int ph = 0; ph < 8; ++ph) {
    const int buf = ph & 1;
    tile_wait();               // phase ph tile landed (TENSORcnt; no-op for others)
    __syncthreads();           // tile visible to all waves; buf^1 readers done
    if (ph + 1 < 8) {
      int kk0 = (ph + 1) * TILEW;
      const bf16* src = (kk0 < 1024) ? (xb_t + kk0) : (Tprev + (kk0 - 1024));
      tile_issue(Abuf[buf ^ 1], src, 1024, tid, wave);
    }
    for (int k0 = 0; k0 < TILEW; k0 += 32) {
      int kk = ph * TILEW + k0;
      v16bf bfrag = load_b_frag(Wcomb, nBase, kk, KDIM, lane);
      v16bf af[4];
#pragma unroll
      for (int mt = 0; mt < 4; ++mt)
        af[mt] = load_a_frag_lds(Abuf[buf], mt * 16 + r0, k0, half);
#pragma unroll
      for (int mt = 0; mt < 4; ++mt)
        acc[mt] = wmma_bf16(af[mt], bfrag, acc[mt]);
    }
  }
  const int n = nBase + (lane & 15);
  const float bias = biasComb[n];
  for (int mt = 0; mt < 4; ++mt)
    for (int j = 0; j < 8; ++j) {
      int m = mt * 16 + (lane >> 4) * 8 + j;
      gates[(size_t)m * NG + n] = acc[mt][j] + bias;
    }
}

// ---------------------------------------------------------------------------
// Step kernel 2: LSTM pointwise (both cells). grid = 512 x 256
// ---------------------------------------------------------------------------
__global__ void k_lstm(const float* __restrict__ gates, float* __restrict__ cF,
                       float* __restrict__ cR, bf16* __restrict__ hF2b,
                       bf16* __restrict__ hR2b) {
  int idx = blockIdx.x * blockDim.x + threadIdx.x;
  int b = idx >> 11, u = idx & 2047;
  int isR = u >> 10, h = u & 1023;
  const float* g = gates + (size_t)b * NG + (size_t)isR * 4096;
  float gi = g[h], gf = g[1024 + h], gg = g[2048 + h], go = g[3072 + h];
  float* c = (isR ? cR : cF) + b * HIDN + h;
  float c2 = sigf(gf) * (*c) + sigf(gi) * tanhf(gg);
  float h2 = sigf(go) * tanhf(c2);
  *c = c2;
  (isR ? hR2b : hF2b)[b * HIDN + h] = to_bf16(h2);
}

// ---------------------------------------------------------------------------
// Step kernel 3: attention logits [64,768]; blocks 0..3 -> F, 4..5 -> R.
// ---------------------------------------------------------------------------
__global__ __launch_bounds__(256) void k_attn(const bf16* __restrict__ hF2b,
                                              const bf16* __restrict__ hR2b,
                                              const bf16* __restrict__ WaFb,
                                              const bf16* __restrict__ WaRb,
                                              const float* __restrict__ WaF_b,
                                              const float* __restrict__ WaR_b,
                                              float* __restrict__ logits) {
  __shared__ bf16 Abuf[2][64 * TILEW];
  const int tid = threadIdx.x, lane = tid & 31, wave = tid >> 5;
  const bool isR = blockIdx.x >= 4;
  const bf16* A = isR ? hR2b : hF2b;
  const bf16* W = isR ? WaRb : WaFb;
  const float* bias = isR ? WaR_b : WaF_b;
  const int nLocal = (isR ? (blockIdx.x - 4) : blockIdx.x) * 128 + wave * 16;
  const int nOut = (isR ? NSYM : 0) + nLocal;
  const int r0 = lane & 15, half = lane >> 4;
  v8f acc[4] = {};

  tile_issue(Abuf[0], A, 1024, tid, wave);
  for (int ph = 0; ph < 4; ++ph) {
    const int buf = ph & 1;
    tile_wait();
    __syncthreads();
    if (ph + 1 < 4)
      tile_issue(Abuf[buf ^ 1], A + (ph + 1) * TILEW, 1024, tid, wave);
    for (int k0 = 0; k0 < TILEW; k0 += 32) {
      int kk = ph * TILEW + k0;
      v16bf bfrag = load_b_frag(W, nLocal, kk, HIDN, lane);
      v16bf af[4];
#pragma unroll
      for (int mt = 0; mt < 4; ++mt)
        af[mt] = load_a_frag_lds(Abuf[buf], mt * 16 + r0, k0, half);
#pragma unroll
      for (int mt = 0; mt < 4; ++mt)
        acc[mt] = wmma_bf16(af[mt], bfrag, acc[mt]);
    }
  }
  const float bv = bias[nLocal + (lane & 15)];
  const int n = nOut + (lane & 15);
  for (int mt = 0; mt < 4; ++mt)
    for (int j = 0; j < 8; ++j) {
      int m = mt * 16 + (lane >> 4) * 8 + j;
      logits[(size_t)m * 768 + n] = acc[mt][j] + bv;
    }
}

// ---------------------------------------------------------------------------
// Step kernel 4: row softmax. blocks 0..63 -> F (len 512), 64..127 -> R (len 256)
// ---------------------------------------------------------------------------
__global__ __launch_bounds__(256) void k_softmax(const float* __restrict__ logits,
                                                 float* __restrict__ outF,
                                                 float* __restrict__ outR,
                                                 bf16* __restrict__ aFb,
                                                 bf16* __restrict__ aRb, int t) {
  __shared__ float red[256];
  const int tid = threadIdx.x;
  const bool isR = blockIdx.x >= 64;
  const int b = isR ? (blockIdx.x - 64) : blockIdx.x;
  const int len = isR ? NROLE : NSYM;
  const float* row = logits + (size_t)b * 768 + (isR ? NSYM : 0);
  float* orow = isR ? (outR + ((size_t)b * Ssz + t) * NROLE)
                    : (outF + ((size_t)b * Ssz + t) * NSYM);
  bf16* brow = isR ? (aRb + (size_t)b * NROLE) : (aFb + (size_t)b * NSYM);

  float m = -1e30f;
  for (int j = tid; j < len; j += 256) m = fmaxf(m, row[j]);
  red[tid] = m; __syncthreads();
  for (int s = 128; s > 0; s >>= 1) { if (tid < s) red[tid] = fmaxf(red[tid], red[tid + s]); __syncthreads(); }
  m = red[0]; __syncthreads();

  float sum = 0.0f;
  for (int j = tid; j < len; j += 256) sum += __expf(row[j] - m);
  red[tid] = sum; __syncthreads();
  for (int s = 128; s > 0; s >>= 1) { if (tid < s) red[tid] += red[tid + s]; __syncthreads(); }
  float inv = 1.0f / red[0];

  for (int j = tid; j < len; j += 256) {
    float v = __expf(row[j] - m) * inv;
    orow[j] = v;
    brow[j] = to_bf16(v);
  }
}

// ---------------------------------------------------------------------------
// Step kernel 5: itemF/itemR via WMMA, T = itemF (x) itemR, carry T -> bf16.
// ---------------------------------------------------------------------------
__global__ __launch_bounds__(256) void k_items(const bf16* __restrict__ aFb,
                                               const bf16* __restrict__ aRb,
                                               const bf16* __restrict__ Fwb,
                                               const bf16* __restrict__ Rwb,
                                               const float* __restrict__ F_b,
                                               const float* __restrict__ R_b,
                                               float* __restrict__ out_iF,
                                               float* __restrict__ out_iR,
                                               float* __restrict__ out_T,
                                               bf16* __restrict__ Tprev, int t) {
  __shared__ float itemF[64 * 32];
  __shared__ float itemR[64 * 32];
  const int tid = threadIdx.x, lane = tid & 31, wave = tid >> 5;
  const bool isR = wave >= 4;
  const int mt = wave & 3;
  const bf16* A = isR ? aRb : aFb;
  const bf16* W = isR ? Rwb : Fwb;
  const float* bias = isR ? R_b : F_b;
  float* item = isR ? itemR : itemF;
  float* oitem = isR ? out_iR : out_iF;
  const int K = isR ? NROLE : NSYM;

  v8f acc[2] = {};
  for (int kk = 0; kk < K; kk += 32) {
    v16bf afrag = load_a_frag_g(A, mt * 16, kk, K, lane);
    v16bf bf0 = load_b_frag(W, 0, kk, K, lane);
    v16bf bf1 = load_b_frag(W, 16, kk, K, lane);
    acc[0] = wmma_bf16(afrag, bf0, acc[0]);
    acc[1] = wmma_bf16(afrag, bf1, acc[1]);
  }
  for (int nt = 0; nt < 2; ++nt) {
    int n = nt * 16 + (lane & 15);
    float bv = bias[n];
    for (int j = 0; j < 8; ++j) {
      int m = mt * 16 + (lane >> 4) * 8 + j;
      float v = acc[nt][j] + bv;
      item[m * 32 + n] = v;
      oitem[((size_t)m * Ssz + t) * 32 + n] = v;
    }
  }
  __syncthreads();
  for (int i = tid; i < 64 * 1024; i += 256) {
    int b = i >> 10, h = i & 1023;
    float v = itemF[b * 32 + (h >> 5)] * itemR[b * 32 + (h & 31)];
    out_T[((size_t)b * Ssz + t) * HIDN + h] = v;
    Tprev[(size_t)b * HIDN + h] = to_bf16(v);
  }
}

// ---------------------------------------------------------------------------
// Host-side orchestration
// ---------------------------------------------------------------------------
static inline size_t alignup(size_t x) { return (x + 255) & ~(size_t)255; }

extern "C" void kernel_launch(void* const* d_in, const int* in_sizes, int n_in,
                              void* d_out, int out_size, void* d_ws, size_t ws_size,
                              hipStream_t stream) {
  const float* x    = (const float*)d_in[0];
  const float* WihF = (const float*)d_in[1];
  const float* WhhF = (const float*)d_in[2];
  const float* bihF = (const float*)d_in[3];
  const float* bhhF = (const float*)d_in[4];
  const float* WihR = (const float*)d_in[5];
  const float* WhhR = (const float*)d_in[6];
  const float* bihR = (const float*)d_in[7];
  const float* bhhR = (const float*)d_in[8];
  const float* WaF_w = (const float*)d_in[9];
  const float* WaF_b = (const float*)d_in[10];
  const float* WaR_w = (const float*)d_in[11];
  const float* WaR_b = (const float*)d_in[12];
  const float* F_w  = (const float*)d_in[13];
  const float* F_b  = (const float*)d_in[14];
  const float* R_w  = (const float*)d_in[15];
  const float* R_b  = (const float*)d_in[16];
  float* out = (float*)d_out;

  char* p = (char*)d_ws;
  size_t off = 0;
  bf16* xb      = (bf16*)(p + off); off = alignup(off + (size_t)Ssz * Bsz * INDIM * sizeof(bf16));
  bf16* Wcomb   = (bf16*)(p + off); off = alignup(off + (size_t)NG * KDIM * sizeof(bf16));
  bf16* WaFb    = (bf16*)(p + off); off = alignup(off + (size_t)NSYM * HIDN * sizeof(bf16));
  bf16* WaRb    = (bf16*)(p + off); off = alignup(off + (size_t)NROLE * HIDN * sizeof(bf16));
  bf16* Fwb     = (bf16*)(p + off); off = alignup(off + (size_t)DSYM * NSYM * sizeof(bf16));
  bf16* Rwb     = (bf16*)(p + off); off = alignup(off + (size_t)DROLE * NROLE * sizeof(bf16));
  float* biasComb = (float*)(p + off); off = alignup(off + (size_t)NG * sizeof(float));
  float* gates  = (float*)(p + off); off = alignup(off + (size_t)Bsz * NG * sizeof(float));
  float* cF     = (float*)(p + off); off = alignup(off + (size_t)Bsz * HIDN * sizeof(float));
  float* cR     = (float*)(p + off); off = alignup(off + (size_t)Bsz * HIDN * sizeof(float));
  bf16* hF2b    = (bf16*)(p + off); off = alignup(off + (size_t)Bsz * HIDN * sizeof(bf16));
  bf16* hR2b    = (bf16*)(p + off); off = alignup(off + (size_t)Bsz * HIDN * sizeof(bf16));
  float* logits = (float*)(p + off); off = alignup(off + (size_t)Bsz * 768 * sizeof(float));
  bf16* aFb     = (bf16*)(p + off); off = alignup(off + (size_t)Bsz * NSYM * sizeof(bf16));
  bf16* aRb     = (bf16*)(p + off); off = alignup(off + (size_t)Bsz * NROLE * sizeof(bf16));
  bf16* Tprev   = (bf16*)(p + off); off = alignup(off + (size_t)Bsz * HIDN * sizeof(bf16));
  (void)ws_size; (void)n_in; (void)in_sizes; (void)out_size;

  const size_t o_T  = 0;
  const size_t o_aF = o_T  + (size_t)Bsz * Ssz * HIDN;
  const size_t o_aR = o_aF + (size_t)Bsz * Ssz * NSYM;
  const size_t o_iF = o_aR + (size_t)Bsz * Ssz * NROLE;
  const size_t o_iR = o_iF + (size_t)Bsz * Ssz * DSYM;

  hipMemsetAsync(cF, 0, (size_t)Bsz * HIDN * sizeof(float), stream);
  hipMemsetAsync(cR, 0, (size_t)Bsz * HIDN * sizeof(float), stream);
  hipMemsetAsync(Tprev, 0, (size_t)Bsz * HIDN * sizeof(bf16), stream);

  k_prep_x<<<4096, 256, 0, stream>>>(x, xb);
  k_prep_w<<<4096, 256, 0, stream>>>(WihF, WhhF, WihR, WhhR, Wcomb);
  k_prep_small<<<2048, 256, 0, stream>>>(bihF, bhhF, bihR, bhhR, WaF_w, WaR_w, F_w, R_w,
                                         biasComb, WaFb, WaRb, Fwb, Rwb);

  for (int t = 0; t < Ssz; ++t) {
    const bf16* xb_t = xb + (size_t)t * Bsz * INDIM;
    k_gates<<<64, 256, 0, stream>>>(xb_t, Tprev, Wcomb, biasComb, gates);
    k_lstm<<<512, 256, 0, stream>>>(gates, cF, cR, hF2b, hR2b);
    k_attn<<<6, 256, 0, stream>>>(hF2b, hR2b, WaFb, WaRb, WaF_b, WaR_b, logits);
    k_softmax<<<128, 256, 0, stream>>>(logits, out + o_aF, out + o_aR, aFb, aRb, t);
    k_items<<<1, 256, 0, stream>>>(aFb, aRb, Fwb, Rwb, F_b, R_b,
                                   out + o_iF, out + o_iR, out + o_T, Tprev, t);
  }
}
